// GKTGNN_3496103379583
// MI455X (gfx1250) — matrix operand order repaired and checked by reading, not compiled, not run on hardware
//
#include <hip/hip_runtime.h>
#include <hip/hip_bf16.h>

typedef __attribute__((ext_vector_type(16))) _Float16 v16h;
typedef __attribute__((ext_vector_type(8)))  _Float16 v8h;
typedef __attribute__((ext_vector_type(8)))  float    v8f;
typedef __attribute__((ext_vector_type(4)))  float    v4f;

#define NNODES 100000
#define DFEAT  128
#define ODIM   12
#define LDSROW 136   // padded LDS row stride in halves (272B; 16B aligned, bank-spread)

// ---------------- utility kernels ----------------

__global__ void k_fill(float* __restrict__ p, float v, long long n) {
  long long i = (long long)blockIdx.x * blockDim.x + threadIdx.x;
  long long s = (long long)gridDim.x * blockDim.x;
  for (; i < n; i += s) p[i] = v;
}

__global__ void k_degree(const int* __restrict__ dst, float* __restrict__ deg, int E) {
  int i = blockIdx.x * blockDim.x + threadIdx.x;
  if (i < E) atomicAdd(&deg[dst[i]], 1.0f);
}

__global__ void k_rsqrt(const float* __restrict__ deg, float* __restrict__ dinv, int n) {
  int i = blockIdx.x * blockDim.x + threadIdx.x;
  if (i < n) {
    float d = deg[i];
    dinv[i] = (d > 0.f) ? rsqrtf(d) : 0.f;
  }
}

// Transpose 128x128 f32 -> f16: Wt[i][j] = (f16)W[j][i]
__global__ void k_transpose128h(const float* __restrict__ W, _Float16* __restrict__ Wt) {
  __shared__ float tile[16][17];
  int bx = blockIdx.x & 7, by = blockIdx.x >> 3;
  int tx = threadIdx.x & 15, ty = threadIdx.x >> 4;
  tile[ty][tx] = W[(by * 16 + ty) * 128 + bx * 16 + tx];
  __syncthreads();
  Wt[(bx * 16 + ty) * 128 + by * 16 + tx] = (_Float16)tile[tx][ty];
}

// Pack Wr [128][12] f32 row-major into padded transposed f16 Wrt [16][128]
__global__ void k_pack_wrt(const float* __restrict__ Wr, _Float16* __restrict__ Wrt) {
  int i = blockIdx.x * blockDim.x + threadIdx.x;
  if (i < 16 * 128) {
    int n = i >> 7, k = i & 127;
    Wrt[i] = (_Float16)((n < ODIM) ? Wr[k * ODIM + n] : 0.f);
  }
}

// ---------------- WMMA GEMM: out = preop(A) @ W (+post_bias) ----------------
// A: [Mrows][128] f32 (optionally row-gathered); Wt: [n_tiles*16][128] f16 (transposed,
// pre-packed). Weights staged global->LDS via CDNA5 async-to-LDS b128 copies; B
// fragments then come from LDS as contiguous 32B runs per lane.
// PRE:     apply relu(a + pre_bias[k]) to A elements (fused previous-layer epilogue)
// POSTB:   add post_bias[col] to outputs
// GATHER:  A rows indexed through gather[]
// COLGUARD: store only cols < out_cols (readout tile padding)
template <bool PRE, bool POSTB, bool GATHER, bool COLGUARD>
__launch_bounds__(256)
__global__ void k_gemm_wmma(const float* __restrict__ A,
                            const _Float16* __restrict__ Wt,
                            float* __restrict__ out,
                            int M_tiles, int Mrows, int n_tiles,
                            const float* __restrict__ pre_bias,
                            const float* __restrict__ post_bias,
                            const int*   __restrict__ gather,
                            int out_cols, int out_ld)
{
  __shared__ _Float16 sw[128 * LDSROW];   // up to 128 weight columns, padded rows

  // --- async stage W^T (f16) into LDS: rows*16 chunks of 16B each ---
  {
    const int nchunks = n_tiles * 16 * 16;         // 16 x b128 per 128-half row
    unsigned long long gbase = (unsigned long long)(size_t)Wt;
    for (int c = threadIdx.x; c < nchunks; c += blockDim.x) {
      int r = c >> 4, seg = c & 15;
      unsigned goff = (unsigned)((r * 128 + seg * 8) * 2);              // global bytes
      unsigned loff = (unsigned)(size_t)&sw[r * LDSROW + seg * 8];      // LDS bytes
      asm volatile("global_load_async_to_lds_b128 %0, %1, %2"
                   :: "v"(loff), "v"(goff), "s"(gbase) : "memory");
    }
    asm volatile("s_wait_asynccnt 0" ::: "memory");
  }
  __syncthreads();

  int wave = blockIdx.x * (blockDim.x >> 5) + (threadIdx.x >> 5);
  if (wave >= M_tiles) return;
  int lane  = threadIdx.x & 31;
  int lhalf = lane >> 4;   // 0: lanes 0-15, 1: lanes 16-31
  int l16   = lane & 15;

  int m0 = wave * 16;
  int rA = m0 + l16;
  if (rA > Mrows - 1) rA = Mrows - 1;         // harmless clamp (M tile-aligned here)
  if (GATHER) rA = gather[rA];
  const float* arow = A + (long long)rA * DFEAT;

  // A fragments for all 4 K-chunks (K=128 = 4 x 32), reused across all n-tiles.
  // A 16x32 layout (matches B): lanes 0-15 hold K 0-15, lanes 16-31 K 16-31.
  v16h afrag[4];
#pragma unroll
  for (int kk = 0; kk < 4; ++kk) {
    int g0 = kk * 32 + (lhalf ? 16 : 0);
#pragma unroll
    for (int g = 0; g < 2; ++g) {
      int kb = g0 + g * 8;
#pragma unroll
      for (int e = 0; e < 8; ++e) {
        float a = arow[kb + e];
        if (PRE) { a += pre_bias[kb + e]; if (a < 0.f) a = 0.f; }
        afrag[kk][g * 8 + e] = (_Float16)a;
      }
    }
  }

  for (int nt = 0; nt < n_tiles; ++nt) {
    int col = nt * 16 + l16;
    const _Float16* wrow = sw + col * LDSROW;   // column of W = padded LDS row

    // Preload all 4 B fragments (8x ds_load_b128), then run the WMMA chain dense.
    v16h bfrag[4];
#pragma unroll
    for (int kk = 0; kk < 4; ++kk) {
      int kb = kk * 32 + (lhalf ? 16 : 0);      // lanes 0-15: K lo16, 16-31: K hi16
      v8h lo = *(const v8h*)(wrow + kb);
      v8h hi = *(const v8h*)(wrow + kb + 8);
#pragma unroll
      for (int e = 0; e < 8; ++e) { bfrag[kk][e] = lo[e]; bfrag[kk][8 + e] = hi[e]; }
    }

    v8f acc = {};
#pragma unroll
    for (int kk = 0; kk < 4; ++kk) {
      acc = __builtin_amdgcn_wmma_f32_16x16x32_f16(
          /*neg_a=*/false, afrag[kk], /*neg_b=*/false, bfrag[kk],
          /*c_mod=*/(short)0, acc, /*reuse_a=*/false, /*reuse_b=*/false);
    }

    bool do_store = !COLGUARD || (col < out_cols);
    if (do_store) {
      float pb = POSTB ? post_bias[col] : 0.f;
#pragma unroll
      for (int r = 0; r < 8; ++r) {
        int row = m0 + r + 8 * lhalf;   // C/D layout: vgpr r -> M = r (+8 for hi half)
        out[(long long)row * out_ld + col] = acc[r] + pb;
      }
    }
  }
}

// ---------------- edge scatter: agg[dst] += t[src] * dinv[src]*dinv[dst] ----------------
// Self-loop "edges" e in [E, E+N) use src=dst=e-E. float4 granularity, atomic adds.
__launch_bounds__(256)
__global__ void k_scatter(const float* __restrict__ t,
                          const int* __restrict__ src, const int* __restrict__ dst,
                          const float* __restrict__ dinv, float* __restrict__ agg,
                          int E, int Nn)
{
  long long total = (long long)(E + Nn) * 32;
  long long i = (long long)blockIdx.x * blockDim.x + threadIdx.x;
  long long stride = (long long)gridDim.x * blockDim.x;
  for (; i < total; i += stride) {
    long long e = i >> 5;
    int c = (int)(i & 31) * 4;
    int s, d;
    if (e < E) {
      s = src[e]; d = dst[e];
      long long en = (i + stride) >> 5;
      if (en < E) { __builtin_prefetch(&src[en], 0, 1); __builtin_prefetch(&dst[en], 0, 1); }
    } else {
      s = d = (int)(e - E);
    }
    float w = dinv[s] * dinv[d];
    const v4f v = *(const v4f*)(t + (long long)s * DFEAT + c);
    float* ap = agg + (long long)d * DFEAT + c;
    atomicAdd(ap + 0, v.x * w);
    atomicAdd(ap + 1, v.y * w);
    atomicAdd(ap + 2, v.z * w);
    atomicAdd(ap + 3, v.w * w);
  }
}

// ---------------- launch ----------------

extern "C" void kernel_launch(void* const* d_in, const int* in_sizes, int n_in,
                              void* d_out, int out_size, void* d_ws, size_t ws_size,
                              hipStream_t stream) {
  const float* x   = (const float*)d_in[0];
  const int*  edge = (const int*)d_in[1];    // [2][E]
  const int*  tgt  = (const int*)d_in[2];    // [NT]
  const float* W1  = (const float*)d_in[3];
  const float* b1  = (const float*)d_in[4];
  const float* W2  = (const float*)d_in[5];
  const float* b2  = (const float*)d_in[6];
  const float* W3  = (const float*)d_in[7];
  const float* b3  = (const float*)d_in[8];
  const float* Wr  = (const float*)d_in[9];
  const float* br  = (const float*)d_in[10];

  const int E  = in_sizes[1] / 2;
  const int NT = in_sizes[2];
  const int* src = edge;
  const int* dst = edge + E;

  // workspace layout: bufA/bufB ping-pong (f32), deg, dinv (f32), packed f16 weights
  float* bufA = (float*)d_ws;                           // N*128 f32
  float* bufB = bufA + (long long)NNODES * DFEAT;       // N*128 f32
  float* deg  = bufB + (long long)NNODES * DFEAT;       // N f32
  float* dinv = deg + NNODES;                           // N f32
  _Float16* Wt1 = (_Float16*)(dinv + NNODES);           // 128*128 f16 (16B aligned)
  _Float16* Wt2 = Wt1 + 128 * 128;
  _Float16* Wt3 = Wt2 + 128 * 128;
  _Float16* Wrt = Wt3 + 128 * 128;                      // 16*128 f16

  // degrees (init 1.0 = self-loop), dinv = rsqrt(deg)
  k_fill<<<(NNODES + 255) / 256, 256, 0, stream>>>(deg, 1.0f, NNODES);
  k_degree<<<(E + 255) / 256, 256, 0, stream>>>(dst, deg, E);
  k_rsqrt<<<(NNODES + 255) / 256, 256, 0, stream>>>(deg, dinv, NNODES);

  // pre-pack weights: transpose + f16 (B fragments load contiguous 32B runs)
  k_transpose128h<<<64, 256, 0, stream>>>(W1, Wt1);
  k_transpose128h<<<64, 256, 0, stream>>>(W2, Wt2);
  k_transpose128h<<<64, 256, 0, stream>>>(W3, Wt3);
  k_pack_wrt<<<(16 * 128 + 255) / 256, 256, 0, stream>>>(Wr, Wrt);

  const int mt = (NNODES + 15) / 16;       // 6250 waves
  const int gB = (mt + 7) / 8;             // 8 waves / 256-thread block
  const long long nfeat = (long long)NNODES * DFEAT;

  // Layer 1: bufB = x @ W1 ; bufA = scatter(bufB)
  k_gemm_wmma<false, false, false, false><<<gB, 256, 0, stream>>>(
      x, Wt1, bufB, mt, NNODES, 8, nullptr, nullptr, nullptr, DFEAT, DFEAT);
  k_fill<<<4096, 256, 0, stream>>>(bufA, 0.0f, nfeat);
  k_scatter<<<8192, 256, 0, stream>>>(bufB, src, dst, dinv, bufA, E, NNODES);

  // Layer 2: bufB = relu(bufA + b1) @ W2 ; bufA = scatter(bufB)
  k_gemm_wmma<true, false, false, false><<<gB, 256, 0, stream>>>(
      bufA, Wt2, bufB, mt, NNODES, 8, b1, nullptr, nullptr, DFEAT, DFEAT);
  k_fill<<<4096, 256, 0, stream>>>(bufA, 0.0f, nfeat);
  k_scatter<<<8192, 256, 0, stream>>>(bufB, src, dst, dinv, bufA, E, NNODES);

  // Layer 3: bufB = relu(bufA + b2) @ W3 ; bufA = scatter(bufB)
  k_gemm_wmma<true, false, false, false><<<gB, 256, 0, stream>>>(
      bufA, Wt3, bufB, mt, NNODES, 8, b2, nullptr, nullptr, DFEAT, DFEAT);
  k_fill<<<4096, 256, 0, stream>>>(bufA, 0.0f, nfeat);
  k_scatter<<<8192, 256, 0, stream>>>(bufB, src, dst, dinv, bufA, E, NNODES);

  // Readout: out[NT,12] = relu(bufA + b3)[tgt] @ Wr + br
  const int mtT = (NT + 15) / 16;          // 625 waves
  k_gemm_wmma<true, true, true, true><<<(mtT + 7) / 8, 256, 0, stream>>>(
      bufA, Wrt, (float*)d_out, mtT, NT, 1, b3, br, tgt, ODIM, ODIM);
}
